// Joiner_6803228197541
// MI455X (gfx1250) — matrix operand order repaired and checked by reading, not compiled, not run on hardware
//
#include <hip/hip_runtime.h>

// Problem constants (from reference)
#define BB 16
#define TT 300
#define UU 30
#define JJ 512   // ENC_DIM = DEC_DIM = JOINER_DIM
#define VV 500   // VOCAB (padded to 512 internally)

typedef __attribute__((ext_vector_type(16))) __bf16 v16bf;
typedef __attribute__((ext_vector_type(8)))  __bf16 v8bf;
typedef __attribute__((ext_vector_type(8)))  float  v8f;
typedef __attribute__((ext_vector_type(4)))  float  f4;

// ---- f32 -> bf16 hi/lo split (round-to-nearest-even), ~f32 accuracy when recombined
__device__ __forceinline__ void splitbf(float x, __bf16& hi, __bf16& lo) {
    unsigned u  = __float_as_uint(x);
    unsigned rh = (u + 0x7FFFu + ((u >> 16) & 1u)) & 0xFFFF0000u;
    hi = __builtin_bit_cast(__bf16, (unsigned short)(rh >> 16));
    float r = x - __uint_as_float(rh);
    unsigned v  = __float_as_uint(r);
    unsigned rl = v + 0x7FFFu + ((v >> 16) & 1u);
    lo = __builtin_bit_cast(__bf16, (unsigned short)(rl >> 16));
}

__device__ __forceinline__ __bf16 bfz() {
    return __builtin_bit_cast(__bf16, (unsigned short)0);
}

__device__ __forceinline__ v8f wmma_bf(v16bf a, v16bf b, v8f c) {
    // D = A(16x32 bf16) x B(32x16 bf16) + C(16x16 f32)
    return __builtin_amdgcn_wmma_f32_16x16x32_bf16(
        /*neg_a=*/false, a, /*neg_b=*/false, b,
        /*c_mod=*/(short)0, c, /*reuse_a=*/false, /*reuse_b=*/false);
}

// Load 16 bf16 as two b128 loads -> v16bf
__device__ __forceinline__ v16bf ld16(const __bf16* p0, const __bf16* p1) {
    v8bf a = *(const v8bf*)p0;
    v8bf b = *(const v8bf*)p1;
    return __builtin_shufflevector(a, b, 0,1,2,3,4,5,6,7,8,9,10,11,12,13,14,15);
}

// ---- Prep: elementwise f32 -> (bf16 hi, bf16 lo)
__global__ __launch_bounds__(256) void split_arr(const float* __restrict__ x,
                                                 __bf16* __restrict__ h,
                                                 __bf16* __restrict__ l, int n) {
    int i = blockIdx.x * 256 + threadIdx.x;
    if (i < n) splitbf(x[i], h[i], l[i]);
}

// ---- Prep: W_out (500x512) -> zero-padded 512x512 hi/lo, plus padded bias
__global__ __launch_bounds__(256) void split_wout(const float* __restrict__ W,
                                                  __bf16* __restrict__ h,
                                                  __bf16* __restrict__ l,
                                                  const float* __restrict__ bias,
                                                  float* __restrict__ biasp) {
    int i = blockIdx.x * 256 + threadIdx.x;   // over 512*512
    int row = i >> 9;
    if (row < VV) splitbf(W[i], h[i], l[i]);
    else { h[i] = bfz(); l[i] = bfz(); }
    if (i < JJ) biasp[i] = (i < VV) ? bias[i] : 0.0f;
}

// ---- Split-precision bf16 WMMA GEMM: C[M,512] = X[M,512] @ W[512,512]^T (+bias)
// X, W given pre-split as bf16 hi/lo. Each wave: 16x32 tile; block (256 thr) : 16x256.
// grid = (2, M/16). OUT_SPLIT: also emit C as bf16 hi/lo (feeds the next GEMM).
template <bool OUT_SPLIT>
__global__ __launch_bounds__(256) void gemm_bf16x2(
    const __bf16* __restrict__ Xh, const __bf16* __restrict__ Xl,
    const __bf16* __restrict__ Wh, const __bf16* __restrict__ Wl,
    const float*  __restrict__ bias,
    float* __restrict__ Cf, __bf16* __restrict__ Ch, __bf16* __restrict__ Cl) {
    constexpr int K = JJ;
    const int lane = threadIdx.x & 31;
    const int wave = threadIdx.x >> 5;
    const int row0 = blockIdx.y << 4;
    const int col0 = (blockIdx.x << 8) | (wave << 5);
    const int m    = lane & 15;
    const bool lo16 = lane < 16;
    const int ha = lo16 ? 0 : 16;   // A: K base within 32-chunk per lane half
    const int hb = lo16 ? 0 : 16;   // B: K base per lane half

    const __bf16* xh  = Xh + (size_t)(row0 + m) * K;
    const __bf16* xl  = Xl + (size_t)(row0 + m) * K;
    const __bf16* w0h = Wh + (size_t)(col0 + m) * K + hb;
    const __bf16* w0l = Wl + (size_t)(col0 + m) * K + hb;
    const __bf16* w1h = Wh + (size_t)(col0 + 16 + m) * K + hb;
    const __bf16* w1l = Wl + (size_t)(col0 + 16 + m) * K + hb;

    v8f acc0 = {}; v8f acc1 = {};
#pragma unroll 2
    for (int kb = 0; kb < K; kb += 32) {
        v16bf ah  = ld16(xh + kb + ha, xh + kb + 8 + ha);
        v16bf al  = ld16(xl + kb + ha, xl + kb + 8 + ha);
        v16bf b0h = ld16(w0h + kb, w0h + kb + 8);
        v16bf b0l = ld16(w0l + kb, w0l + kb + 8);
        v16bf b1h = ld16(w1h + kb, w1h + kb + 8);
        v16bf b1l = ld16(w1l + kb, w1l + kb + 8);
        // x*w ~= ah*bh + ah*bl + al*bh  (drop lo*lo), f32 accumulate
        acc0 = wmma_bf(al, b0h, acc0);
        acc0 = wmma_bf(ah, b0l, acc0);
        acc0 = wmma_bf(ah, b0h, acc0);
        acc1 = wmma_bf(al, b1h, acc1);
        acc1 = wmma_bf(ah, b1l, acc1);
        acc1 = wmma_bf(ah, b1h, acc1);
    }
    const float bv0 = bias ? bias[col0 + m]      : 0.0f;
    const float bv1 = bias ? bias[col0 + 16 + m] : 0.0f;
    const int rb = row0 + (lo16 ? 0 : 8);
#pragma unroll
    for (int r = 0; r < 8; ++r) {
        size_t i0 = (size_t)(rb + r) * JJ + col0 + m;
        size_t i1 = i0 + 16;
        float v0 = acc0[r] + bv0;
        float v1 = acc1[r] + bv1;
        if constexpr (OUT_SPLIT) {
            splitbf(v0, Ch[i0], Cl[i0]);
            splitbf(v1, Ch[i1], Cl[i1]);
        } else {
            Cf[i0] = v0;
            Cf[i1] = v1;
        }
    }
}

// ---- Final broadcast add: out[b,t,u,0:500] = A[b*T+t, 0:500] + Dv[b*U+u, 0:500]
// One block per (b,t); A row cached in registers; 288MB streamed with NT stores.
__global__ __launch_bounds__(128) void joint_bcast(const float* __restrict__ A,
                                                   const float* __restrict__ Dv,
                                                   float* __restrict__ out) {
    const int row = blockIdx.x;       // b*T + t
    const int b   = row / TT;
    const int tid = threadIdx.x;
    if (tid >= VV / 4) return;        // 125 active lanes, 500 = 125*4
    f4 a4 = *(const f4*)(A + (size_t)row * JJ + tid * 4);
    const float* dbase = Dv + (size_t)b * UU * JJ + tid * 4;
    float* obase = out + (size_t)row * UU * VV + (size_t)tid * 4;
#pragma unroll 5
    for (int u = 0; u < UU; ++u) {
        f4 d4 = *(const f4*)(dbase + (size_t)u * JJ);
        f4 o  = a4 + d4;
        __builtin_nontemporal_store(o, (f4*)(obase + (size_t)u * VV));
    }
}

extern "C" void kernel_launch(void* const* d_in, const int* in_sizes, int n_in,
                              void* d_out, int out_size, void* d_ws, size_t ws_size,
                              hipStream_t stream) {
    const float* enc  = (const float*)d_in[0];  // (B,T,512)
    const float* dec  = (const float*)d_in[1];  // (B,U,512)
    const float* Wenc = (const float*)d_in[2];  // (512,512)
    const float* benc = (const float*)d_in[3];  // (512)
    const float* Wdec = (const float*)d_in[4];  // (512,512)
    const float* bdec = (const float*)d_in[5];  // (512)
    const float* Wout = (const float*)d_in[6];  // (500,512)
    const float* bout = (const float*)d_in[7];  // (500)
    float* out = (float*)d_out;                 // (B,T,U,500)

    const size_t NE = (size_t)BB * TT * JJ;   // 2,457,600
    const size_t ND = (size_t)BB * UU * JJ;   //   245,760
    const size_t NW = (size_t)JJ * JJ;        //   262,144

    char* p = (char*)d_ws;
    auto alloc = [&](size_t bytes) {
        char* r = p;
        p += (bytes + 255) & ~(size_t)255;
        return r;
    };
    __bf16* ench  = (__bf16*)alloc(NE * 2);
    __bf16* encl  = (__bf16*)alloc(NE * 2);
    __bf16* dech  = (__bf16*)alloc(ND * 2);
    __bf16* decl  = (__bf16*)alloc(ND * 2);
    __bf16* wench = (__bf16*)alloc(NW * 2);
    __bf16* wencl = (__bf16*)alloc(NW * 2);
    __bf16* wdech = (__bf16*)alloc(NW * 2);
    __bf16* wdecl = (__bf16*)alloc(NW * 2);
    __bf16* wouth = (__bf16*)alloc(NW * 2);
    __bf16* woutl = (__bf16*)alloc(NW * 2);
    float*  boutp = (float*)alloc(JJ * 4);
    __bf16* eph   = (__bf16*)alloc(NE * 2);   // enc projection (bf16 hi/lo)
    __bf16* epl   = (__bf16*)alloc(NE * 2);
    __bf16* dph   = (__bf16*)alloc(ND * 2);   // dec projection (bf16 hi/lo)
    __bf16* dpl   = (__bf16*)alloc(ND * 2);
    float*  Abuf  = (float*)alloc(NE * 4);    // A  = enc_p @ Wout^T + b_out (4800x512)
    float*  Dbuf  = (float*)alloc(ND * 4);    // Dv = dec_p @ Wout^T         (480x512)

    // 1) Split all GEMM operands into bf16 hi/lo (and pad W_out/b_out to 512)
    split_arr<<<(int)((NE + 255) / 256), 256, 0, stream>>>(enc,  ench,  encl,  (int)NE);
    split_arr<<<(int)((ND + 255) / 256), 256, 0, stream>>>(dec,  dech,  decl,  (int)ND);
    split_arr<<<(int)((NW + 255) / 256), 256, 0, stream>>>(Wenc, wench, wencl, (int)NW);
    split_arr<<<(int)((NW + 255) / 256), 256, 0, stream>>>(Wdec, wdech, wdecl, (int)NW);
    split_wout<<<(int)(NW / 256), 256, 0, stream>>>(Wout, wouth, woutl, bout, boutp);

    // 2) Projections (emit bf16 hi/lo for the next GEMM)
    gemm_bf16x2<true ><<<dim3(2, (BB * TT) / 16), 256, 0, stream>>>(
        ench, encl, wench, wencl, benc, nullptr, eph, epl);
    gemm_bf16x2<true ><<<dim3(2, (BB * UU) / 16), 256, 0, stream>>>(
        dech, decl, wdech, wdecl, bdec, nullptr, dph, dpl);

    // 3) Vocab projections of each factor (f32 out)
    gemm_bf16x2<false><<<dim3(2, (BB * TT) / 16), 256, 0, stream>>>(
        eph, epl, wouth, woutl, boutp, Abuf, nullptr, nullptr);
    gemm_bf16x2<false><<<dim3(2, (BB * UU) / 16), 256, 0, stream>>>(
        dph, dpl, wouth, woutl, nullptr, Dbuf, nullptr, nullptr);

    // 4) Bandwidth-bound broadcast add -> 288 MB output
    joint_bcast<<<BB * TT, 128, 0, stream>>>(Abuf, Dbuf, out);
}
